// MultiTaskGCN_18330920419573
// MI455X (gfx1250) — matrix operand order repaired and checked by reading, not compiled, not run on hardware
//
#include <hip/hip_runtime.h>

#define N_NODES 100000
#define N_EDGES 1600000
#define M_TILES 6250   // 100000 / 16, exact

typedef __attribute__((ext_vector_type(16))) __bf16 v16bf;
typedef __attribute__((ext_vector_type(8)))  float  v8f;

__device__ __forceinline__ unsigned short f32_bf16(float f) {
  unsigned int x = __float_as_uint(f);
  unsigned int r = (x + 0x7FFFu + ((x >> 16) & 1u)) >> 16;
  return (unsigned short)r;
}

__global__ void k_fill(float* p, float v, size_t n) {
  size_t i = (size_t)blockIdx.x * blockDim.x + threadIdx.x;
  if (i < n) p[i] = v;
}

__global__ void k_deg(const int* __restrict__ dst, float* __restrict__ deg, int e) {
  int i = blockIdx.x * blockDim.x + threadIdx.x;
  if (i < e) unsafeAtomicAdd(&deg[dst[i]], 1.0f);
}

__global__ void k_rsqrt(float* p, int n) {
  int i = blockIdx.x * blockDim.x + threadIdx.x;
  if (i < n) p[i] = rsqrtf(p[i]);
}

__global__ void k_f32_to_bf16(const float* __restrict__ in, unsigned short* __restrict__ out, size_t n) {
  size_t i = (size_t)blockIdx.x * blockDim.x + threadIdx.x;
  if (i < n) out[i] = f32_bf16(in[i]);
}

// Pack f32 W[K,NO] into WMMA B-fragment order (bf16 pairs), so each lane's
// fragment for (tileN, kstep) is 8 contiguous dwords:
//   out[(((tileN*KSTEPS + ks)*32 + lane)*8 + j)] = {W[kb,col], W[kb+1,col]}
//   kb = ks*32 + 2(j&3) + (j>>2)*16 + (lane>>4)*8, col = tileN*16 + (lane&15)
__global__ void k_packW(const float* __restrict__ W, unsigned int* __restrict__ out,
                        int K, int NO) {
  int t = blockIdx.x * blockDim.x + threadIdx.x;
  int total = (K / 2) * NO;
  if (t >= total) return;
  int ksteps = K / 32;
  int j     = t & 7;
  int lane  = (t >> 3) & 31;
  int grp   = t >> 8;             // tileN*KSTEPS + ks
  int ks    = grp % ksteps;
  int tileN = grp / ksteps;
  int kb  = ks * 32 + (j & 3) * 2 + (j >> 2) * 16 + ((lane >> 4) * 8);
  int col = tileN * 16 + (lane & 15);
  unsigned int lo = f32_bf16(W[(size_t)kb * NO + col]);
  unsigned int hi = f32_bf16(W[(size_t)(kb + 1) * NO + col]);
  out[t] = lo | (hi << 16);
}

// C[M,NO] = A[M,K] * B[K,NO]. A bf16 row-major, Bp pre-packed fragments.
// One wave computes a full 16-row x NO strip (A fragments reused across tiles).
template <int K, int NO>
__global__ void k_gemm_bf16(const unsigned short* __restrict__ A,
                            const uint4* __restrict__ Bp,
                            float* __restrict__ C) {
  constexpr int KSTEPS = K / 32;
  constexpr int NTILES = NO / 16;
  int wave  = threadIdx.x >> 5;
  int lane  = threadIdx.x & 31;
  int tileM = blockIdx.x * (blockDim.x >> 5) + wave;
  if (tileM >= M_TILES) return;            // wave-uniform: EXEC all-1 for WMMA
  int laneHi = lane >> 4;
  int lane15 = lane & 15;
  int row    = tileM * 16 + lane15;

  union Frag { v16bf v; uint4 q[2]; };
  Frag a, b;
  v8f c[NTILES];
#pragma unroll
  for (int t = 0; t < NTILES; ++t) c[t] = (v8f){};

  // A row has K bf16 = K/8 uint4; fragment = uint4 {ks*4+laneHi, ks*4+2+laneHi}
  const uint4* arow = (const uint4*)(A + (size_t)row * K);
#pragma unroll
  for (int ks = 0; ks < KSTEPS; ++ks) {
    a.q[0] = arow[ks * 4 + laneHi];
    a.q[1] = arow[ks * 4 + 2 + laneHi];
#pragma unroll
    for (int t = 0; t < NTILES; ++t) {
      const uint4* bp = Bp + ((size_t)(t * KSTEPS + ks) * 32 + lane) * 2;
      b.q[0] = bp[0];
      b.q[1] = bp[1];
      c[t] = __builtin_amdgcn_wmma_f32_16x16x32_bf16(false, a.v, false, b.v,
                                                     (short)0, c[t], false, false);
    }
  }
  // D 16x16 f32: VGPR r -> M = r + laneHi*8, N = lane15
#pragma unroll
  for (int t = 0; t < NTILES; ++t) {
    float* crow = C + (size_t)(tileM * 16 + laneHi * 8) * NO + t * 16 + lane15;
#pragma unroll
    for (int r = 0; r < 8; ++r) crow[(size_t)r * NO] = c[t][r];
  }
}

// agg[dst] += hW[src] * dis[src]*dis[dst]; blockIdx.y = float4 chunk index
__global__ void k_scatter(const int* __restrict__ src, const int* __restrict__ dst,
                          const float* __restrict__ dis, const float* __restrict__ hW,
                          float* __restrict__ agg, int outF) {
  int e = blockIdx.x * blockDim.x + threadIdx.x;
  if (e >= N_EDGES) return;
  int cchunk = blockIdx.y;
  int s = src[e], d = dst[e];
  float nm = dis[s] * dis[d];
  float4 v = *((const float4*)(hW + (size_t)s * outF) + cchunk);
  float* o = agg + (size_t)d * outF + (cchunk << 2);
  unsafeAtomicAdd(o + 0, v.x * nm);
  unsafeAtomicAdd(o + 1, v.y * nm);
  unsafeAtomicAdd(o + 2, v.z * nm);
  unsafeAtomicAdd(o + 3, v.w * nm);
}

// out = relu(agg + hW*dis^2 + bias); optional bf16 copy for next layer's GEMM
__global__ void k_finalize(const float* __restrict__ hW, const float* __restrict__ agg,
                           const float* __restrict__ dis, const float* __restrict__ bias,
                           float* __restrict__ outF32, unsigned short* outB16, int outF) {
  int node = blockIdx.x;
  int f = threadIdx.x;
  size_t i = (size_t)node * outF + f;
  float di = dis[node];
  float v = agg[i] + hW[i] * di * di + bias[f];
  v = fmaxf(v, 0.0f);
  outF32[i] = v;
  if (outB16) outB16[i] = f32_bf16(v);
}

__global__ void k_heads(const float* __restrict__ emb,
                        const float* __restrict__ Wt1, const float* __restrict__ bt1,
                        const float* __restrict__ Wt2, const float* __restrict__ bt2,
                        const float* __restrict__ Wp,  const float* __restrict__ bp,
                        const float* __restrict__ pos,
                        float* __restrict__ turnover, float* __restrict__ pref) {
  int i = blockIdx.x * blockDim.x + threadIdx.x;
  if (i >= N_NODES) return;
  float e[32];
#pragma unroll
  for (int k = 0; k < 32; ++k) e[k] = emb[(size_t)i * 32 + k];

  float acc = bt2[0];
  for (int j = 0; j < 16; ++j) {
    float t = bt1[j];
#pragma unroll
    for (int k = 0; k < 32; ++k) t += e[k] * Wt1[k * 16 + j];
    acc += fmaxf(t, 0.0f) * Wt2[j];
  }
  turnover[i] = acc;

  float pr[32];
  for (int j = 0; j < 32; ++j) {
    float p = bp[j];
#pragma unroll
    for (int k = 0; k < 32; ++k) p += e[k] * Wp[k * 32 + j];
    pr[j] = p;
  }
  for (int p = 0; p < 7; ++p) {
    float s = 0.0f;
#pragma unroll
    for (int k = 0; k < 32; ++k) s += pr[k] * pos[p * 32 + k];
    pref[(size_t)i * 7 + p] = s;
  }
}

extern "C" void kernel_launch(void* const* d_in, const int* in_sizes, int n_in,
                              void* d_out, int out_size, void* d_ws, size_t ws_size,
                              hipStream_t stream) {
  const float* x   = (const float*)d_in[0];
  const int*   ei  = (const int*)d_in[1];
  const int* srcI  = ei;
  const int* dstI  = ei + N_EDGES;
  const float* W1  = (const float*)d_in[2];
  const float* b1  = (const float*)d_in[3];
  const float* W2  = (const float*)d_in[4];
  const float* b2  = (const float*)d_in[5];
  const float* W3  = (const float*)d_in[6];
  const float* b3  = (const float*)d_in[7];
  const float* Wt1 = (const float*)d_in[8];
  const float* bt1 = (const float*)d_in[9];
  const float* Wt2 = (const float*)d_in[10];
  const float* bt2 = (const float*)d_in[11];
  const float* Wp  = (const float*)d_in[12];
  const float* bp  = (const float*)d_in[13];
  const float* pos = (const float*)d_in[14];

  char* ws = (char*)d_ws;
  float*        dis = (float*)ws;                                  // 400 KB
  unsigned int* pw1 = (unsigned int*)(ws + (512u << 10));          // 8192 dwords
  unsigned int* pw2 = pw1 + 128 * 128 / 2;                         // 4096 dwords
  unsigned int* pw3 = pw2 + 128 * 64 / 2;                          // 1024 dwords
  float*          bufA = (float*)(ws + ((size_t)1 << 20));         // hW, 51.2 MB max
  float*          bufB = (float*)(ws + ((size_t)53 << 20));        // agg, 51.2 MB max
  unsigned short* h16  = (unsigned short*)(ws + ((size_t)106 << 20)); // bf16 acts

  dim3 b256(256);

  // degrees -> dis = 1/sqrt(deg+1)
  k_fill<<<(N_NODES + 255) / 256, b256, 0, stream>>>(dis, 1.0f, (size_t)N_NODES);
  k_deg<<<(N_EDGES + 255) / 256, b256, 0, stream>>>(dstI, dis, N_EDGES);
  k_rsqrt<<<(N_NODES + 255) / 256, b256, 0, stream>>>(dis, N_NODES);

  // bf16 conversions / weight packing
  size_t nx = (size_t)N_NODES * 128;
  k_f32_to_bf16<<<(unsigned)((nx + 255) / 256), b256, 0, stream>>>(x, h16, nx);
  k_packW<<<(128 * 128 / 2 + 255) / 256, b256, 0, stream>>>(W1, pw1, 128, 128);
  k_packW<<<(128 * 64 / 2 + 255) / 256, b256, 0, stream>>>(W2, pw2, 128, 64);
  k_packW<<<(64 * 32 / 2 + 255) / 256, b256, 0, stream>>>(W3, pw3, 64, 32);

  float* emb  = (float*)d_out;                      // [N,32]
  float* turn = emb + (size_t)N_NODES * 32;         // [N]
  float* pref = turn + N_NODES;                     // [N,7]

  const unsigned gmWaves = (M_TILES + 7) / 8;       // 8 waves / block of 256

  // ---- Layer 1: 128 -> 128 ----
  {
    k_gemm_bf16<128, 128><<<gmWaves, b256, 0, stream>>>(h16, (const uint4*)pw1, bufA);
    size_t tot = (size_t)N_NODES * 128;
    k_fill<<<(unsigned)((tot + 255) / 256), b256, 0, stream>>>(bufB, 0.0f, tot);
    dim3 gs((N_EDGES + 255) / 256, 128 / 4);
    k_scatter<<<gs, b256, 0, stream>>>(srcI, dstI, dis, bufA, bufB, 128);
    k_finalize<<<N_NODES, dim3(128), 0, stream>>>(bufA, bufB, dis, b1, bufB, h16, 128);
  }
  // ---- Layer 2: 128 -> 64 ----
  {
    k_gemm_bf16<128, 64><<<gmWaves, b256, 0, stream>>>(h16, (const uint4*)pw2, bufA);
    size_t tot = (size_t)N_NODES * 64;
    k_fill<<<(unsigned)((tot + 255) / 256), b256, 0, stream>>>(bufB, 0.0f, tot);
    dim3 gs((N_EDGES + 255) / 256, 64 / 4);
    k_scatter<<<gs, b256, 0, stream>>>(srcI, dstI, dis, bufA, bufB, 64);
    k_finalize<<<N_NODES, dim3(64), 0, stream>>>(bufA, bufB, dis, b2, bufB, h16, 64);
  }
  // ---- Layer 3: 64 -> 32 ----
  {
    k_gemm_bf16<64, 32><<<gmWaves, b256, 0, stream>>>(h16, (const uint4*)pw3, bufA);
    size_t tot = (size_t)N_NODES * 32;
    k_fill<<<(unsigned)((tot + 255) / 256), b256, 0, stream>>>(bufB, 0.0f, tot);
    dim3 gs((N_EDGES + 255) / 256, 32 / 4);
    k_scatter<<<gs, b256, 0, stream>>>(srcI, dstI, dis, bufA, bufB, 32);
    k_finalize<<<N_NODES, dim3(32), 0, stream>>>(bufA, bufB, dis, b3, emb, nullptr, 32);
  }

  k_heads<<<(N_NODES + 255) / 256, b256, 0, stream>>>(emb, Wt1, bt1, Wt2, bt2,
                                                      Wp, bp, pos, turn, pref);
}
